// AttentionPoolBuildingBlock_7361573945691
// MI455X (gfx1250) — matrix operand order repaired and checked by reading, not compiled, not run on hardware
//
#include <hip/hip_runtime.h>
#include <hip/hip_bf16.h>

// Problem constants (match reference)
#define N_ATOMS  100000
#define B_BB     10000
#define E_EDGES  320000
#define D_DIM    256
#define H_HEADS  8
#define HD_DIM   32
#define DIST_DIM 64
#define K2       (D_DIM + DIST_DIM)   // 320

#define G_STEP   (10.0f / 63.0f)               // linspace(0,10,64) spacing
#define G_COEFF  (-0.5f / (G_STEP * G_STEP))   // -0.5/step^2
#define INV_SQRT_HD 0.17677669529663687f       // 1/sqrt(32)

typedef __attribute__((ext_vector_type(16))) _Float16 v16h;
typedef __attribute__((ext_vector_type(8)))  _Float16 v8h;
typedef __attribute__((ext_vector_type(4)))  _Float16 v4h;
typedef __attribute__((ext_vector_type(8)))  float    v8f;

#define SHFL16(lo, hi) __builtin_shufflevector(lo, hi, 0,1,2,3,4,5,6,7,8,9,10,11,12,13,14,15)

// ---------------------------------------------------------------- utilities

__global__ __launch_bounds__(256) void fill_f32(float* p, float v, long n) {
    long i = (long)blockIdx.x * 256 + threadIdx.x;
    if (i < n) p[i] = v;
}

// W[K][256] (f32, row-major) -> WT[256][K] (f16, column-of-W contiguous)
__global__ __launch_bounds__(256)
void transpose_to_f16(const float* __restrict__ W, _Float16* __restrict__ WT, int K) {
    int idx = blockIdx.x * 256 + threadIdx.x;   // over 256*K, output-linear
    if (idx < 256 * K) {
        int n = idx / K;
        int k = idx - n * K;
        WT[idx] = (_Float16)W[(long)k * 256 + n];
    }
}

__device__ inline void atomicMaxFloat(float* addr, float val) {
    int* ai = (int*)addr;
    int old = __float_as_int(*addr);
    while (val > __int_as_float(old)) {
        int assumed = old;
        old = atomicCAS(ai, assumed, __float_as_int(val));
        if (old == assumed) break;
    }
}

// ------------------------------------------------- segment sums over atoms

__global__ __launch_bounds__(256)
void atom_pos_count(const float* __restrict__ coords, const int* __restrict__ bb_vec,
                    float* __restrict__ counts, float* __restrict__ possum) {
    int i = blockIdx.x * 256 + threadIdx.x;
    if (i < N_ATOMS) {
        int b = bb_vec[i];
        atomicAdd(&counts[b], 1.0f);
        atomicAdd(&possum[b * 3 + 0], coords[i * 3 + 0]);
        atomicAdd(&possum[b * 3 + 1], coords[i * 3 + 1]);
        atomicAdd(&possum[b * 3 + 2], coords[i * 3 + 2]);
    }
}

__global__ __launch_bounds__(256)
void atom_feat_scatter(const float* __restrict__ atom_feats, const int* __restrict__ bb_vec,
                       float* __restrict__ featsum) {
    long idx = (long)blockIdx.x * 256 + threadIdx.x;   // over N*256
    int a = (int)(idx >> 8);
    int d = (int)(idx & 255);
    if (a < N_ATOMS)
        atomicAdd(&featsum[(long)bb_vec[a] * 256 + d], atom_feats[idx]);
}

__global__ __launch_bounds__(256)
void bb_finalize(const float* __restrict__ counts, const float* __restrict__ possum,
                 float* __restrict__ bbpos) {
    int i = blockIdx.x * 256 + threadIdx.x;
    if (i < B_BB * 3) bbpos[i] = possum[i] / fmaxf(counts[i / 3], 1.0f);
}

__global__ __launch_bounds__(256)
void edge_dist_kernel(const float* __restrict__ coords, const float* __restrict__ bbpos,
                      const int* __restrict__ src, const int* __restrict__ tgt,
                      float* __restrict__ dist) {
    int e = blockIdx.x * 256 + threadIdx.x;
    if (e < E_EDGES) {
        int s = src[e], t = tgt[e];
        float dx = coords[s * 3 + 0] - bbpos[t * 3 + 0];
        float dy = coords[s * 3 + 1] - bbpos[t * 3 + 1];
        float dz = coords[s * 3 + 2] - bbpos[t * 3 + 2];
        dist[e] = sqrtf(dx * dx + dy * dy + dz * dz);
    }
}

// --------------------------------------------------------- WMMA GEMM (K=256, N=256)
// Y[M,256] = X[M,256] @ W + bias, W pre-transposed as WT[256][256] f16.
// Block = 256 threads = 8 waves; block tile 32 rows x 256 cols.
// A frag: 2x ds_load_b128 from row-major LDS tile; B frag: one 32B global load
// per tile from WT (L2-resident). No barriers inside the K loop.

#define XP256 264   // padded row length (16B-aligned rows, spread banks)

__global__ __launch_bounds__(256)
void gemm256(const float* __restrict__ X, const _Float16* __restrict__ WT,
             const float* __restrict__ bias, float* __restrict__ Y, int M) {
    __shared__ _Float16 sX[32][XP256];
    const int tid  = threadIdx.x;
    const int lane = tid & 31;
    const int wave = tid >> 5;
    const int half = lane >> 4;
    const int r    = lane & 15;
    const int rowTile  = wave >> 2;   // 0..1
    const int colGroup = wave & 3;    // 0..3
    const int m0 = blockIdx.x * 32;

    // stage full X tile [32][256] as f16 (float4 loads, v4h stores)
    for (int i = tid * 4; i < 32 * 256; i += 1024) {
        int row = i >> 8;
        int c   = i & 255;
        int gr  = m0 + row;
        float4 v = make_float4(0.f, 0.f, 0.f, 0.f);
        if (gr < M) v = *(const float4*)(X + (long)gr * 256 + c);
        v4h h;
        h[0] = (_Float16)v.x; h[1] = (_Float16)v.y;
        h[2] = (_Float16)v.z; h[3] = (_Float16)v.w;
        *(v4h*)(&sX[row][c]) = h;
    }
    __syncthreads();

    v8f acc[4];
#pragma unroll
    for (int j = 0; j < 4; ++j)
#pragma unroll
        for (int i = 0; i < 8; ++i) acc[j][i] = 0.0f;

    const _Float16* ap  = &sX[rowTile * 16 + r][0];
    const _Float16* bp0 = WT + (long)(colGroup * 64 + r) * 256;

    for (int k0 = 0; k0 < 256; k0 += 32) {
        v8h alo = *(const v8h*)(ap + k0 + 16 * half);
        v8h ahi = *(const v8h*)(ap + k0 + 16 * half + 8);
        v16h a = SHFL16(alo, ahi);
#pragma unroll
        for (int j = 0; j < 4; ++j) {
            v16h b = *(const v16h*)(bp0 + (long)j * 16 * 256 + k0 + 16 * half);
            acc[j] = __builtin_amdgcn_wmma_f32_16x16x32_f16(
                false, a, false, b, (short)0, acc[j], false, false);
        }
    }

#pragma unroll
    for (int j = 0; j < 4; ++j) {
        int col = colGroup * 64 + j * 16 + r;
        float bv = bias[col];
#pragma unroll
        for (int i = 0; i < 8; ++i) {
            int gr = m0 + rowTile * 16 + i + 8 * half;  // C/D row = i + 8*half
            if (gr < M) Y[(long)gr * 256 + col] = acc[j][i] + bv;
        }
    }
}

// --------------------------------------------- fused edge pass A: K + logits

#define XPK2 328    // K2=320 padded row length

__global__ __launch_bounds__(256)
void edge_k_logits(const float* __restrict__ atom_feats,
                   const int* __restrict__ src, const int* __restrict__ tgt,
                   const float* __restrict__ dist,
                   const _Float16* __restrict__ WkT, const float* __restrict__ bk,
                   const float* __restrict__ q,
                   float* __restrict__ attn, float* __restrict__ segmax) {
    __shared__ _Float16 sX[32][XPK2];
    __shared__ float    sK[32][256];
    __shared__ int      s_src[32];
    __shared__ int      s_tgt[32];
    __shared__ float    s_dist[32];

    const int tid  = threadIdx.x;
    const int lane = tid & 31;
    const int wave = tid >> 5;
    const int half = lane >> 4;
    const int r    = lane & 15;
    const int rowTile  = wave >> 2;
    const int colGroup = wave & 3;
    const int e0 = blockIdx.x * 32;

    if (tid < 32) {
        int e = e0 + tid;
        s_src[tid]  = (e < E_EDGES) ? src[e] : 0;
        s_tgt[tid]  = (e < E_EDGES) ? tgt[e] : 0;
        s_dist[tid] = (e < E_EDGES) ? dist[e] : 0.0f;
    }
    __syncthreads();

    // stage full [32][320] input tile: gathered feats + gaussian smear cols
    for (int i = tid * 4; i < 32 * K2; i += 1024) {
        int row = i / K2;
        int c   = i - row * K2;       // multiple of 4; chunk never straddles 256
        v4h h;
        if (c < D_DIM) {
            float4 v = *(const float4*)(atom_feats + (long)s_src[row] * D_DIM + c);
            h[0] = (_Float16)v.x; h[1] = (_Float16)v.y;
            h[2] = (_Float16)v.z; h[3] = (_Float16)v.w;
        } else {
            float base = s_dist[row];
#pragma unroll
            for (int j = 0; j < 4; ++j) {
                float off = G_STEP * (float)(c + j - D_DIM);
                float dd  = base - off;
                h[j] = (_Float16)expf(G_COEFF * dd * dd);
            }
        }
        *(v4h*)(&sX[row][c]) = h;
    }
    __syncthreads();

    v8f acc[4];
#pragma unroll
    for (int j = 0; j < 4; ++j)
#pragma unroll
        for (int i = 0; i < 8; ++i) acc[j][i] = 0.0f;

    const _Float16* ap  = &sX[rowTile * 16 + r][0];
    const _Float16* bp0 = WkT + (long)(colGroup * 64 + r) * K2;

    for (int k0 = 0; k0 < K2; k0 += 32) {
        v8h alo = *(const v8h*)(ap + k0 + 16 * half);
        v8h ahi = *(const v8h*)(ap + k0 + 16 * half + 8);
        v16h a = SHFL16(alo, ahi);
#pragma unroll
        for (int j = 0; j < 4; ++j) {
            v16h b = *(const v16h*)(bp0 + (long)j * 16 * K2 + k0 + 16 * half);
            acc[j] = __builtin_amdgcn_wmma_f32_16x16x32_f16(
                false, a, false, b, (short)0, acc[j], false, false);
        }
    }

    // spill K tile (+bias) to LDS
#pragma unroll
    for (int j = 0; j < 4; ++j) {
        int col = colGroup * 64 + j * 16 + r;
        float bv = bk[col];
#pragma unroll
        for (int i = 0; i < 8; ++i)
            sK[rowTile * 16 + i + 8 * half][col] = acc[j][i] + bv;
    }
    __syncthreads();

    // per-(edge, head) logits vs gathered q[tgt]
    int row = tid >> 3;
    int h   = tid & 7;
    int e   = e0 + row;
    if (e < E_EDGES) {
        int b = s_tgt[row];
        const float* qrow = q + (long)b * 256 + h * HD_DIM;
        float s = 0.0f;
#pragma unroll
        for (int d = 0; d < HD_DIM; ++d) s += qrow[d] * sK[row][h * HD_DIM + d];
        float logit = s * INV_SQRT_HD;
        attn[(long)e * H_HEADS + h] = logit;
        atomicMaxFloat(&segmax[b * H_HEADS + h], logit);
    }
}

// ------------------------------------------ segment softmax normalize pass

__global__ __launch_bounds__(256)
void softmax_norm(const int* __restrict__ tgt, const float* __restrict__ segmax,
                  float* __restrict__ attn, float* __restrict__ denom) {
    long idx = (long)blockIdx.x * 256 + threadIdx.x;   // over E*H
    if (idx < (long)E_EDGES * H_HEADS) {
        int e = (int)(idx >> 3);
        int h = (int)(idx & 7);
        int b = tgt[e];
        float x = expf(attn[idx] - segmax[b * H_HEADS + h]);
        attn[idx] = x;
        atomicAdd(&denom[b * H_HEADS + h], x);
    }
}

// ---------------------------------- fused edge pass B: V + weighted scatter

__global__ __launch_bounds__(256)
void edge_v_scatter(const float* __restrict__ atom_feats,
                    const int* __restrict__ src, const int* __restrict__ tgt,
                    const float* __restrict__ dist,
                    const _Float16* __restrict__ WvT, const float* __restrict__ bv,
                    const float* __restrict__ ex, const float* __restrict__ denom,
                    float* __restrict__ outacc) {
    __shared__ _Float16 sX[32][XPK2];
    __shared__ float    sV[32][256];
    __shared__ int      s_src[32];
    __shared__ int      s_tgt[32];
    __shared__ float    s_dist[32];

    const int tid  = threadIdx.x;
    const int lane = tid & 31;
    const int wave = tid >> 5;
    const int half = lane >> 4;
    const int r    = lane & 15;
    const int rowTile  = wave >> 2;
    const int colGroup = wave & 3;
    const int e0 = blockIdx.x * 32;

    if (tid < 32) {
        int e = e0 + tid;
        s_src[tid]  = (e < E_EDGES) ? src[e] : 0;
        s_tgt[tid]  = (e < E_EDGES) ? tgt[e] : 0;
        s_dist[tid] = (e < E_EDGES) ? dist[e] : 0.0f;
    }
    __syncthreads();

    for (int i = tid * 4; i < 32 * K2; i += 1024) {
        int row = i / K2;
        int c   = i - row * K2;
        v4h h;
        if (c < D_DIM) {
            float4 v = *(const float4*)(atom_feats + (long)s_src[row] * D_DIM + c);
            h[0] = (_Float16)v.x; h[1] = (_Float16)v.y;
            h[2] = (_Float16)v.z; h[3] = (_Float16)v.w;
        } else {
            float base = s_dist[row];
#pragma unroll
            for (int j = 0; j < 4; ++j) {
                float off = G_STEP * (float)(c + j - D_DIM);
                float dd  = base - off;
                h[j] = (_Float16)expf(G_COEFF * dd * dd);
            }
        }
        *(v4h*)(&sX[row][c]) = h;
    }
    __syncthreads();

    v8f acc[4];
#pragma unroll
    for (int j = 0; j < 4; ++j)
#pragma unroll
        for (int i = 0; i < 8; ++i) acc[j][i] = 0.0f;

    const _Float16* ap  = &sX[rowTile * 16 + r][0];
    const _Float16* bp0 = WvT + (long)(colGroup * 64 + r) * K2;

    for (int k0 = 0; k0 < K2; k0 += 32) {
        v8h alo = *(const v8h*)(ap + k0 + 16 * half);
        v8h ahi = *(const v8h*)(ap + k0 + 16 * half + 8);
        v16h a = SHFL16(alo, ahi);
#pragma unroll
        for (int j = 0; j < 4; ++j) {
            v16h b = *(const v16h*)(bp0 + (long)j * 16 * K2 + k0 + 16 * half);
            acc[j] = __builtin_amdgcn_wmma_f32_16x16x32_f16(
                false, a, false, b, (short)0, acc[j], false, false);
        }
    }

#pragma unroll
    for (int j = 0; j < 4; ++j) {
        int col = colGroup * 64 + j * 16 + r;
        float bb = bv[col];
#pragma unroll
        for (int i = 0; i < 8; ++i)
            sV[rowTile * 16 + i + 8 * half][col] = acc[j][i] + bb;
    }
    __syncthreads();

    // weighted scatter-sum into per-bb accumulator
    for (int i = tid; i < 32 * 256; i += 256) {
        int row = i >> 8;
        int d   = i & 255;
        int h   = d >> 5;
        int e   = e0 + row;
        if (e < E_EDGES) {
            int b = s_tgt[row];
            float w = ex[(long)e * H_HEADS + h] / (denom[b * H_HEADS + h] + 1e-16f);
            atomicAdd(&outacc[(long)b * 256 + d], w * sV[row][d]);
        }
    }
}

// ----------------------------------------------------------------- launcher

extern "C" void kernel_launch(void* const* d_in, const int* in_sizes, int n_in,
                              void* d_out, int out_size, void* d_ws, size_t ws_size,
                              hipStream_t stream) {
    const float* atom_feats = (const float*)d_in[0];
    const float* coords     = (const float*)d_in[1];
    const float* Wq         = (const float*)d_in[2];
    const float* bq         = (const float*)d_in[3];
    const float* Wk         = (const float*)d_in[4];
    const float* bk         = (const float*)d_in[5];
    const float* Wv         = (const float*)d_in[6];
    const float* bvp        = (const float*)d_in[7];
    const float* Wo         = (const float*)d_in[8];
    const float* bo         = (const float*)d_in[9];
    const int*   bb_vec     = (const int*)d_in[10];
    const int*   src        = (const int*)d_in[11];
    const int*   tgt        = (const int*)d_in[12];
    float* out = (float*)d_out;

    // ---- carve workspace (floats) ----
    float* ws = (float*)d_ws;
    long off = 0;
    float* counts  = ws + off; off += B_BB;
    float* possum  = ws + off; off += (long)B_BB * 3;
    float* bbpos   = ws + off; off += (long)B_BB * 3;
    float* featsum = ws + off; off += (long)B_BB * 256;
    float* qbuf    = ws + off; off += (long)B_BB * 256;
    float* segmax  = ws + off; off += (long)B_BB * H_HEADS;
    float* denom   = ws + off; off += (long)B_BB * H_HEADS;
    float* outacc  = ws + off; off += (long)B_BB * 256;
    float* attn    = ws + off; off += (long)E_EDGES * H_HEADS;
    float* dist    = ws + off; off += E_EDGES;
    _Float16* WqT = (_Float16*)(ws + off); off += (256 * 256) / 2;
    _Float16* WkT = (_Float16*)(ws + off); off += (K2 * 256) / 2;
    _Float16* WvT = (_Float16*)(ws + off); off += (K2 * 256) / 2;
    _Float16* WoT = (_Float16*)(ws + off); off += (256 * 256) / 2;
    (void)ws_size; (void)in_sizes; (void)n_in; (void)out_size;

    auto cdiv = [](long a, long b) { return (int)((a + b - 1) / b); };

    // ---- init accumulators (ws is not re-zeroed by harness) ----
    fill_f32<<<cdiv((long)B_BB * 4, 256), 256, 0, stream>>>(counts, 0.0f, (long)B_BB * 4);
    fill_f32<<<cdiv((long)B_BB * 256, 256), 256, 0, stream>>>(featsum, 0.0f, (long)B_BB * 256);
    fill_f32<<<cdiv((long)B_BB * H_HEADS, 256), 256, 0, stream>>>(segmax, -3.0e38f, (long)B_BB * H_HEADS);
    fill_f32<<<cdiv((long)B_BB * H_HEADS, 256), 256, 0, stream>>>(denom, 0.0f, (long)B_BB * H_HEADS);
    fill_f32<<<cdiv((long)B_BB * 256, 256), 256, 0, stream>>>(outacc, 0.0f, (long)B_BB * 256);

    // ---- weights: f32 [K][256] -> f16 transposed [256][K] ----
    transpose_to_f16<<<cdiv(256 * 256, 256), 256, 0, stream>>>(Wq, WqT, 256);
    transpose_to_f16<<<cdiv(256 * K2, 256), 256, 0, stream>>>(Wk, WkT, K2);
    transpose_to_f16<<<cdiv(256 * K2, 256), 256, 0, stream>>>(Wv, WvT, K2);
    transpose_to_f16<<<cdiv(256 * 256, 256), 256, 0, stream>>>(Wo, WoT, 256);

    // ---- segment sums / centers ----
    atom_pos_count<<<cdiv(N_ATOMS, 256), 256, 0, stream>>>(coords, bb_vec, counts, possum);
    atom_feat_scatter<<<N_ATOMS, 256, 0, stream>>>(atom_feats, bb_vec, featsum);
    bb_finalize<<<cdiv((long)B_BB * 3, 256), 256, 0, stream>>>(counts, possum, bbpos);

    // ---- per-edge distances ----
    edge_dist_kernel<<<cdiv(E_EDGES, 256), 256, 0, stream>>>(coords, bbpos, src, tgt, dist);

    // ---- Q = featsum @ Wq + bq  (WMMA) ----
    gemm256<<<cdiv(B_BB, 32), 256, 0, stream>>>(featsum, WqT, bq, qbuf, B_BB);

    // ---- edge pass A: K GEMM + logits + segment max (WMMA) ----
    edge_k_logits<<<cdiv(E_EDGES, 32), 256, 0, stream>>>(
        atom_feats, src, tgt, dist, WkT, bk, qbuf, attn, segmax);

    // ---- segment softmax normalization ----
    softmax_norm<<<cdiv((long)E_EDGES * H_HEADS, 256), 256, 0, stream>>>(tgt, segmax, attn, denom);

    // ---- edge pass B: V GEMM + weighted scatter (WMMA) ----
    edge_v_scatter<<<cdiv(E_EDGES, 32), 256, 0, stream>>>(
        atom_feats, src, tgt, dist, WvT, bvp, attn, denom, outacc);

    // ---- out = outacc @ Wo + bo  (WMMA) ----
    gemm256<<<cdiv(B_BB, 32), 256, 0, stream>>>(outacc, WoT, bo, out, B_BB);
}
